// Leaf_21277267985152
// MI455X (gfx1250) — matrix operand order repaired and checked by reading, not compiled, not run on hardware
//
#include <hip/hip_runtime.h>
#include <hip/hip_bf16.h>
#include <math.h>

typedef __attribute__((ext_vector_type(16))) _Float16 v16h;
typedef __attribute__((ext_vector_type(8)))  float    v8f;

#define LDSTR 132   // f16 LDS row stride (128 + pad, avoids bank conflicts)

// ---------------------------------------------------------------------------
// WMMA helpers (v_wmma_f32_16x16x32_f16, wave32).
// A-frag layout (16x32 f16, ISA 7.12.2): lane L holds row M=L&15;
//   element t -> K = (t&8)*2 + (t&7) + (L>=16 ? 8 : 0)  (+32*kb)
// B-frag layout (32x16 f16): lane L holds col N=L&15;
//   element t -> K = (t&8)*2 + (t&7) + (L>=16 ? 8 : 0)                (+32*kb)
// C/D (16x16 f32): lane L holds col N=L&15; vgpr r -> M = r + (L>=16 ? 8 : 0)
// ---------------------------------------------------------------------------
__device__ __forceinline__ void load_afrags(const _Float16* src, int lane, v16h af[4]) {
  const int M   = lane & 15;
  const int kHi = (lane >= 16) ? 8 : 0;
#pragma unroll
  for (int kb = 0; kb < 4; ++kb) {
    v16h a;
#pragma unroll
    for (int t = 0; t < 16; ++t) {
      int K = kb * 32 + ((t & 8) << 1) + (t & 7) + kHi;
      a[t] = src[M * LDSTR + K];
    }
    af[kb] = a;
  }
}

__device__ __forceinline__ v8f tile16(const v16h af[4], const float* __restrict__ W,
                                      const float* __restrict__ bias,
                                      int Nout, int ct, int lane) {
  const int  col   = ct * 16 + (lane & 15);
  const int  kOff  = (lane >= 16) ? 8 : 0;
  const bool valid = (col < Nout);
  const float bv   = valid ? bias[col] : 0.f;
  v8f acc = {bv, bv, bv, bv, bv, bv, bv, bv};
#pragma unroll
  for (int kb = 0; kb < 4; ++kb) {
    v16h bfr;
#pragma unroll
    for (int t = 0; t < 16; ++t) {
      int K = kb * 32 + ((t & 8) << 1) + (t & 7) + kOff;
      bfr[t] = valid ? (_Float16)W[(size_t)K * Nout + col] : (_Float16)0.f;
    }
    acc = __builtin_amdgcn_wmma_f32_16x16x32_f16(false, af[kb], false, bfr,
                                                 (short)0, acc, false, false);
  }
  return acc;
}

__device__ __forceinline__ void hidden_layer(const _Float16* inL, _Float16* outL,
                                             const float* __restrict__ W,
                                             const float* __restrict__ bias, int lane) {
  v16h af[4];
  load_afrags(inL, lane, af);
  const int colN = lane & 15;
  const int rHi  = (lane >= 16) ? 8 : 0;
#pragma unroll
  for (int ct = 0; ct < 8; ++ct) {
    v8f acc = tile16(af, W, bias, 128, ct, lane);
#pragma unroll
    for (int r = 0; r < 8; ++r) {
      float v = acc[r];
      v = fmaxf(v, 0.2f * v);                 // leaky_relu(0.2)
      outL[(r + rHi) * LDSTR + ct * 16 + colN] = (_Float16)v;
    }
  }
}

// ---------------------------------------------------------------------------
// Kernel A: the three MLP stacks (WMMA) + gumbel argmax.
// One block per 16 rows of the flattened [512,128] input; 8 waves.
// Wave0: time-stack, Wave1: amp-stack, Wave2: sel hidden layers (shared LDS),
// then ALL waves split the 128 column tiles of the 128->2048 sel head and
// fuse gumbel-noise + running argmax (forward value of hard gumbel-softmax).
// ---------------------------------------------------------------------------
__global__ __launch_bounds__(256)
void mlp_select_kernel(const float* __restrict__ x, const float* __restrict__ times,
                       const float* __restrict__ gu,
                       const float* tw0, const float* tb0, const float* tw1, const float* tb1,
                       const float* tw2, const float* tb2,
                       const float* sw0, const float* sb0, const float* sw1, const float* sb1,
                       const float* sw2, const float* sb2,
                       const float* aw0, const float* ab0, const float* aw1, const float* ab1,
                       const float* aw2, const float* ab2,
                       float* __restrict__ wtime, float* __restrict__ wamp,
                       int* __restrict__ widx) {
  __shared__ _Float16 xl[16 * LDSTR];
  __shared__ _Float16 h2s[16 * LDSTR];
  __shared__ _Float16 bufA[3][16 * LDSTR];
  __shared__ _Float16 bufB[2][16 * LDSTR];
  __shared__ float redV[8][16];
  __shared__ int   redI[8][16];

  const int tid   = threadIdx.x;
  const int lane  = tid & 31;
  const int wv    = tid >> 5;
  const int rbase = blockIdx.x * 16;

  for (int i = tid; i < 16 * 128; i += 256) {
    int r = i >> 7, k = i & 127;
    xl[r * LDSTR + k] = (_Float16)x[(size_t)(rbase + r) * 128 + k];
  }
  __syncthreads();

  if (wv == 0) {                                  // time stack -> sigmoid + times
    hidden_layer(xl, bufA[0], tw0, tb0, lane);
    hidden_layer(bufA[0], bufB[0], tw1, tb1, lane);
    v16h af[4]; load_afrags(bufB[0], lane, af);
    v8f acc = tile16(af, tw2, tb2, 1, 0, lane);
    if ((lane & 15) == 0) {
      int rHi = (lane >= 16) ? 8 : 0;
      for (int r = 0; r < 8; ++r) {
        int row = rbase + rHi + r;
        wtime[row] = 1.f / (1.f + __expf(-acc[r])) + times[row];
      }
    }
  } else if (wv == 1) {                           // amp stack -> square
    hidden_layer(xl, bufA[1], aw0, ab0, lane);
    hidden_layer(bufA[1], bufB[1], aw1, ab1, lane);
    v16h af[4]; load_afrags(bufB[1], lane, af);
    v8f acc = tile16(af, aw2, ab2, 1, 0, lane);
    if ((lane & 15) == 0) {
      int rHi = (lane >= 16) ? 8 : 0;
      for (int r = 0; r < 8; ++r) {
        int row = rbase + rHi + r;
        wamp[row] = acc[r] * acc[r];
      }
    }
  } else if (wv == 2) {                           // sel hidden layers (shared)
    hidden_layer(xl, bufA[2], sw0, sb0, lane);
    hidden_layer(bufA[2], h2s, sw1, sb1, lane);
  }
  __syncthreads();

  // sel head 128->2048, fused gumbel + argmax (forward of hard gumbel-softmax)
  v16h af[4]; load_afrags(h2s, lane, af);
  float best[8]; int bidx[8];
#pragma unroll
  for (int r = 0; r < 8; ++r) { best[r] = -1e30f; bidx[r] = 0x7fffffff; }
  const int rHi = (lane >= 16) ? 8 : 0;
  for (int ti = 0; ti < 16; ++ti) {
    int ct  = wv + ti * 8;                        // this wave's column tiles (ascending)
    v8f acc = tile16(af, sw2, sb2, 2048, ct, lane);
    int col = ct * 16 + (lane & 15);
#pragma unroll
    for (int r = 0; r < 8; ++r) {
      int   row = rbase + rHi + r;
      float u   = gu[(size_t)row * 2048 + col];
      float g   = -__logf(-__logf(u + 1e-10f) + 1e-10f);
      float v   = acc[r] + g;
      if (v > best[r]) { best[r] = v; bidx[r] = col; }   // keep-first on tie
    }
  }
  // cross-lane argmax inside each 16-lane half (ties -> smaller index, = jnp.argmax)
#pragma unroll
  for (int off = 1; off < 16; off <<= 1) {
#pragma unroll
    for (int r = 0; r < 8; ++r) {
      float ov = __shfl_xor(best[r], off, 32);
      int   oi = __shfl_xor(bidx[r], off, 32);
      if (ov > best[r] || (ov == best[r] && oi < bidx[r])) { best[r] = ov; bidx[r] = oi; }
    }
  }
  if ((lane & 15) == 0) {
    for (int r = 0; r < 8; ++r) { redV[wv][rHi + r] = best[r]; redI[wv][rHi + r] = bidx[r]; }
  }
  __syncthreads();
  if (tid < 16) {
    float bv = redV[0][tid]; int bi = redI[0][tid];
    for (int w = 1; w < 8; ++w) {
      float ov = redV[w][tid]; int oi = redI[w][tid];
      if (ov > bv || (ov == bv && oi < bi)) { bv = ov; bi = oi; }
    }
    widx[rbase + tid] = bi;
  }
}

// ---------------------------------------------------------------------------
// Kernel B: inverse norms of atom rows: 1/(||atoms[n]|| + 1e-8)
// ---------------------------------------------------------------------------
__global__ __launch_bounds__(128)
void atom_norm_kernel(const float* __restrict__ atoms, float* __restrict__ invn) {
  __shared__ float red[4];
  const int row = blockIdx.x;
  const int tid = threadIdx.x;
  const float* a = atoms + (size_t)row * 512;
  float s = 0.f;
  for (int j = tid; j < 512; j += 128) { float v = a[j]; s = fmaf(v, v, s); }
#pragma unroll
  for (int off = 16; off >= 1; off >>= 1) s += __shfl_xor(s, off, 32);
  if ((tid & 31) == 0) red[tid >> 5] = s;
  __syncthreads();
  if (tid == 0) {
    float t = red[0] + red[1] + red[2] + red[3];
    invn[row] = 1.f / (sqrtf(t) + 1e-8f);
  }
}

// ---------------------------------------------------------------------------
// Kernel C: exact closed-form of the rfft/phase/irfft shift as a circular
// convolution with a Dirichlet kernel, applied only over the 512 nonzero atom
// samples, summed over E. N = 98304, n_coeffs = 49153.
//   c[d] = (1/N) * (-1)^d * ( cos(N*theta/2) - sin(pi*d/N + b1r)/sin(pi*d/N - theta/2) )
// with b1r = (N-1)*theta/2 mod 2pi (reduced in f64 once per event).
// Grid: (64 m-tiles of 512, 8 batches) x 256 threads; e-loop inside (no atomics).
// ---------------------------------------------------------------------------
__global__ __launch_bounds__(256)
void shift_accum_kernel(const float* __restrict__ atoms,
                        const float* __restrict__ wtime, const float* __restrict__ wamp,
                        const int* __restrict__ widx, const float* __restrict__ invn,
                        float* __restrict__ out) {
  __shared__ float tab[1024];
  __shared__ float at[512];
  const int tid  = threadIdx.x;
  const int b    = blockIdx.y;
  const int base = blockIdx.x * 512;
  const double PI   = 3.14159265358979323846;
  const float  invN = 1.0f / 98304.0f;
  float acc0 = 0.f, acc1 = 0.f;

  for (int e = 0; e < 64; ++e) {
    const int   row = b * 64 + e;
    const float tm  = wtime[row];
    const float amp = wamp[row];
    const int   idx = widx[row];
    const float* arow = atoms + (size_t)idx * 512;
    __builtin_prefetch(arow, 0, 0);               // global_prefetch_b8

    const double theta = 2.0 * PI * ((double)tm * 16384.0) / 49153.0;
    const double b1r   = fmod(49151.5 * theta, 2.0 * PI);
    const float  cNyq  = (float)cos(fmod(49152.0 * theta, 2.0 * PI));
    const double th2   = 0.5 * theta;

    for (int t = tid; t < 1024; t += 256) {       // kernel table slice for this tile
      int    d  = base - 512 + t;
      double du = (double)d * (PI / 98304.0);
      float  s1 = __sinf((float)(du + b1r));
      float  s2 = __sinf((float)(du - th2));
      if (fabsf(s2) < 1e-30f) s2 = (s2 < 0.f) ? -1e-30f : 1e-30f;
      float val = (cNyq - s1 / s2) * invN;
      tab[t] = (d & 1) ? -val : val;
    }
    const float scale = invn[idx] * amp;
    for (int j = tid; j < 512; j += 256) at[j] = arow[j] * scale;
    __syncthreads();

#pragma unroll 4
    for (int j = 0; j < 512; ++j) {               // 512-tap FIR, 2 outputs/thread
      float aj = at[j];                           // LDS broadcast
      int   t0 = tid + 512 - j;                   // lane-stride-1: conflict-free
      acc0 = fmaf(aj, tab[t0 - 256], acc0);       // m = base + tid
      acc1 = fmaf(aj, tab[t0],       acc1);       // m = base + 256 + tid
    }
    __syncthreads();
  }
  size_t o = (size_t)b * 32768 + base + tid;
  out[o]       = acc0;
  out[o + 256] = acc1;
}

// ---------------------------------------------------------------------------
extern "C" void kernel_launch(void* const* d_in, const int* in_sizes, int n_in,
                              void* d_out, int out_size, void* d_ws, size_t ws_size,
                              hipStream_t stream) {
  const float* x     = (const float*)d_in[0];
  const float* times = (const float*)d_in[1];
  const float* gu    = (const float*)d_in[2];
  const float* atoms = (const float*)d_in[3];
  const float* tw0 = (const float*)d_in[4],  *tb0 = (const float*)d_in[5];
  const float* tw1 = (const float*)d_in[6],  *tb1 = (const float*)d_in[7];
  const float* tw2 = (const float*)d_in[8],  *tb2 = (const float*)d_in[9];
  const float* sw0 = (const float*)d_in[10], *sb0 = (const float*)d_in[11];
  const float* sw1 = (const float*)d_in[12], *sb1 = (const float*)d_in[13];
  const float* sw2 = (const float*)d_in[14], *sb2 = (const float*)d_in[15];
  const float* aw0 = (const float*)d_in[16], *ab0 = (const float*)d_in[17];
  const float* aw1 = (const float*)d_in[18], *ab1 = (const float*)d_in[19];
  const float* aw2 = (const float*)d_in[20], *ab2 = (const float*)d_in[21];

  char*  ws    = (char*)d_ws;
  float* wtime = (float*)(ws);            // 512 f32
  float* wamp  = (float*)(ws + 2048);     // 512 f32
  int*   widx  = (int*)  (ws + 4096);     // 512 i32
  float* winv  = (float*)(ws + 6144);     // 2048 f32   (total 14336 B)

  mlp_select_kernel<<<32, 256, 0, stream>>>(
      x, times, gu,
      tw0, tb0, tw1, tb1, tw2, tb2,
      sw0, sb0, sw1, sb1, sw2, sb2,
      aw0, ab0, aw1, ab1, aw2, ab2,
      wtime, wamp, widx);

  atom_norm_kernel<<<2048, 128, 0, stream>>>(atoms, winv);

  shift_accum_kernel<<<dim3(64, 8), 256, 0, stream>>>(
      atoms, wtime, wamp, widx, winv, (float*)d_out);
}
